// Head_42262478193015
// MI455X (gfx1250) — matrix operand order repaired and checked by reading, not compiled, not run on hardware
//
#include <hip/hip_runtime.h>
#include <hip/hip_bf16.h>

// Problem constants (from reference): B=128, T=512, C=128, H=64
#define B_ 128
#define T_ 512
#define C_ 128
#define H_ 64

typedef __attribute__((ext_vector_type(16))) __bf16 v16bf;
typedef __attribute__((ext_vector_type(8)))  __bf16 v8bf;
typedef __attribute__((ext_vector_type(8)))  float  v8f;

// workspace layout in __bf16 elements
#define WGT_OFF 0                         // WgT [C][C]   (d-major, K contiguous)
#define WQT_OFF 16384                     // WqT [H][C]
#define WKT_OFF 24576                     // WkT [H][C]
#define WVT_OFF 32768                     // WvT [H][C]
#define Q_OFF   40960                     // Q  [B][T][H]
#define K_OFF   (40960 + 4194304)         // K  [B][T][H]
#define V_OFF   (40960 + 2*4194304)       // Vt [B][H][T]
// total = 2*(40960 + 3*4194304) bytes ~= 25.3 MB of d_ws

__device__ __forceinline__ v16bf cat16(v8bf a, v8bf b) {
  v16bf r;
#pragma unroll
  for (int i = 0; i < 8; ++i) { r[i] = a[i]; r[i + 8] = b[i]; }
  return r;
}

__device__ __forceinline__ v8f wmma_bf16(v16bf a, v16bf b, v8f c) {
  // D = A(16x32 bf16) * B(32x16 bf16) + C(16x16 f32)
  return __builtin_amdgcn_wmma_f32_16x16x32_bf16(false, a, false, b, (short)0, c,
                                                 false, false);
}

// ---------------------------------------------------------------------------
// Kernel 1: convert weights f32 -> bf16, transposed to K-major so WMMA
// B-operand reads are 32B contiguous. 16384 threads.
// ---------------------------------------------------------------------------
__global__ void cvt_weights(const float* __restrict__ Wg,
                            const float* __restrict__ Wk,
                            const float* __restrict__ Wq,
                            const float* __restrict__ Wv,
                            __bf16* __restrict__ ws) {
  int i = blockIdx.x * 256 + threadIdx.x;
  if (i < C_ * C_) {
    int c = i / C_, d = i % C_;
    ws[WGT_OFF + d * C_ + c] = (__bf16)Wg[i];
  }
  if (i < C_ * H_) {
    int c = i / H_, h = i % H_;
    ws[WQT_OFF + h * C_ + c] = (__bf16)Wq[i];
    ws[WKT_OFF + h * C_ + c] = (__bf16)Wk[i];
    ws[WVT_OFF + h * C_ + c] = (__bf16)Wv[i];
  }
}

// ---------------------------------------------------------------------------
// Kernel 2: gate + QKV projections.
// Grid: B * (T/128) blocks, 256 threads (8 waves). Wave w owns 16 rows.
//   xg = x * sigmoid(x@Wg + bg);  q=xg@Wq, k=xg@Wk (-> [B,T,H] bf16),
//   v=xg@Wv stored transposed (-> [B,H,T] bf16).
// 80 v_wmma per wave.
// ---------------------------------------------------------------------------
__global__ __launch_bounds__(256) void gate_qkv(const float* __restrict__ x,
                                                const float* __restrict__ bg,
                                                __bf16* __restrict__ ws) {
  __shared__ __bf16 xs[8][16][136];  // per-wave 16x128 tile, +8 pad (272B rows)
  const int w = threadIdx.x >> 5, lane = threadIdx.x & 31;
  const int hl = lane >> 4, lo = lane & 15;
  const int b = blockIdx.x >> 2, qt = blockIdx.x & 3;
  const int t0 = qt * 128 + w * 16;

  // stage x rows -> bf16 LDS (each wave loads only its own 16 rows)
#pragma unroll
  for (int r = 0; r < 16; ++r) {
    const float4 f = *(const float4*)(x + ((size_t)b * T_ + (t0 + r)) * C_ + lane * 4);
    xs[w][r][lane * 4 + 0] = (__bf16)f.x;
    xs[w][r][lane * 4 + 1] = (__bf16)f.y;
    xs[w][r][lane * 4 + 2] = (__bf16)f.z;
    xs[w][r][lane * 4 + 3] = (__bf16)f.w;
  }
  asm volatile("s_wait_dscnt 0" ::: "memory");

  // A-layout (16x32 bf16): lane holds row lo; K contiguous 16*hl..16*hl+15
  v16bf xa[4];
#pragma unroll
  for (int c = 0; c < 4; ++c) {
    const __bf16* p = &xs[w][lo][32 * c + 16 * hl];
    xa[c] = cat16(*(const v8bf*)p, *(const v8bf*)(p + 8));
  }

  const __bf16* wgT = ws + WGT_OFF;
  // gate GEMM: 8 n-tiles of 16 columns, K=128 (4 WMMA each)
#pragma unroll
  for (int nt = 0; nt < 8; ++nt) {
    v8f acc = {};
#pragma unroll
    for (int c = 0; c < 4; ++c) {
      const __bf16* wp = wgT + (nt * 16 + lo) * C_ + 32 * c + 16 * hl;
      acc = wmma_bf16(xa[c], cat16(*(const v8bf*)wp, *(const v8bf*)(wp + 8)), acc);
    }
    const float bias = bg[nt * 16 + lo];
#pragma unroll
    for (int j = 0; j < 8; ++j) {
      const int m = j + 8 * hl;  // C-layout row
      float g = acc[j] + bias;
      g = 1.f / (1.f + __expf(-g));
      const float xv = (float)xs[w][m][nt * 16 + lo];
      xs[w][m][nt * 16 + lo] = (__bf16)(xv * g);  // overwrite x with xg
    }
  }
  asm volatile("s_wait_dscnt 0" ::: "memory");

  v16bf ga[4];
#pragma unroll
  for (int c = 0; c < 4; ++c) {
    const __bf16* p = &xs[w][lo][32 * c + 16 * hl];
    ga[c] = cat16(*(const v8bf*)p, *(const v8bf*)(p + 8));
  }

  // q / k / v projections (H=64 -> 4 n-tiles, K=128 -> 4 WMMA each)
#pragma unroll
  for (int mtx = 0; mtx < 3; ++mtx) {
    const __bf16* wT = (mtx == 0) ? ws + WQT_OFF
                     : (mtx == 1) ? ws + WKT_OFF
                                  : ws + WVT_OFF;
#pragma unroll
    for (int nt = 0; nt < 4; ++nt) {
      v8f acc = {};
#pragma unroll
      for (int c = 0; c < 4; ++c) {
        const __bf16* wp = wT + (nt * 16 + lo) * C_ + 32 * c + 16 * hl;
        acc = wmma_bf16(ga[c], cat16(*(const v8bf*)wp, *(const v8bf*)(wp + 8)), acc);
      }
      const int hcol = nt * 16 + lo;
#pragma unroll
      for (int j = 0; j < 8; ++j) {
        const int m = j + 8 * hl;
        const int t = t0 + m;
        const __bf16 val = (__bf16)acc[j];
        if (mtx == 0)      ws[Q_OFF + ((size_t)b * T_ + t) * H_ + hcol] = val;
        else if (mtx == 1) ws[K_OFF + ((size_t)b * T_ + t) * H_ + hcol] = val;
        else               ws[V_OFF + ((size_t)b * H_ + hcol) * T_ + t] = val;
      }
    }
  }
}

// ---------------------------------------------------------------------------
// Kernel 3: causal flash attention.
// Grid: B * (T/128) blocks, 256 threads (8 waves). Wave w owns 16 queries.
// K/V staged in LDS in 128-key chunks (uses the large CDNA5 LDS); online
// softmax in f32; 8 v_wmma per 32-key subchunk per wave.
// ---------------------------------------------------------------------------
__global__ __launch_bounds__(256) void flash_attn(const __bf16* __restrict__ ws,
                                                  float* __restrict__ out) {
  __shared__ __bf16 sK[128 * 72];   // [key][H+8pad]   (144B rows)
  __shared__ __bf16 sV[64 * 136];   // [h][128key+8pad] (272B rows)
  __shared__ __bf16 sP[8][16 * 32]; // per-wave P tile, C-layout -> A-layout relay
  const int w = threadIdx.x >> 5, lane = threadIdx.x & 31;
  const int hl = lane >> 4, lo = lane & 15;
  const int b = blockIdx.x >> 2, qt = blockIdx.x & 3;
  const int qb = qt * 128, qlo = qb + w * 16;

  const __bf16* Qg = ws + Q_OFF + (size_t)b * T_ * H_;
  const __bf16* Kg = ws + K_OFF + (size_t)b * T_ * H_;
  const __bf16* Vg = ws + V_OFF + (size_t)b * H_ * T_;

  // q tile in A-layout registers (K = H = 64 -> two K=32 chunks)
  v16bf qa[2];
#pragma unroll
  for (int c = 0; c < 2; ++c) {
    const __bf16* p = Qg + (qlo + lo) * H_ + 32 * c + 16 * hl;
    qa[c] = cat16(*(const v8bf*)p, *(const v8bf*)(p + 8));
  }

  v8f O[4] = {};                  // 16 x 64 output accumulator (4 n-tiles)
  float mrun[8], lrun[8];
#pragma unroll
  for (int j = 0; j < 8; ++j) { mrun[j] = -1e30f; lrun[j] = 0.f; }

  const int nchunk = qt + 1;      // causal: keys up to qb+128
  for (int ch = 0; ch < nchunk; ++ch) {
    const int k0 = ch * 128;
    __syncthreads();              // previous chunk's LDS reads done
    // stage K chunk: 128 rows x 64 bf16 -> 1024 x 16B transfers
#pragma unroll
    for (int it = 0; it < 4; ++it) {
      const int idx = threadIdx.x + 256 * it;
      const int r = idx >> 3, g = idx & 7;
      *(uint4*)&sK[r * 72 + g * 8] = *(const uint4*)(Kg + (size_t)(k0 + r) * H_ + g * 8);
    }
    // stage Vt chunk: 64 rows x 128 bf16
#pragma unroll
    for (int it = 0; it < 4; ++it) {
      const int idx = threadIdx.x + 256 * it;
      const int h = idx >> 4, g = idx & 15;
      *(uint4*)&sV[h * 136 + g * 8] = *(const uint4*)(Vg + (size_t)h * T_ + k0 + g * 8);
    }
    __syncthreads();

#pragma unroll
    for (int s = 0; s < 4; ++s) {
      const int s0 = k0 + 32 * s;
      if (s0 <= qlo + 15) {       // this subchunk intersects causal region
        const int kl = 32 * s;
        // S = q @ k^T  (16 queries x 32 keys), f32 accum
        v8f S0 = {}, S1 = {};
#pragma unroll
        for (int c = 0; c < 2; ++c) {
          const __bf16* kp0 = &sK[(kl + lo) * 72 + 32 * c + 16 * hl];
          S0 = wmma_bf16(qa[c], cat16(*(const v8bf*)kp0, *(const v8bf*)(kp0 + 8)), S0);
          const __bf16* kp1 = &sK[(kl + 16 + lo) * 72 + 32 * c + 16 * hl];
          S1 = wmma_bf16(qa[c], cat16(*(const v8bf*)kp1, *(const v8bf*)(kp1 + 8)), S1);
        }
        const float scale = 0.08838834764831845f;  // C^-0.5
        // online softmax; row m = j + 8*hl spans the 16 lanes of this half
#pragma unroll
        for (int j = 0; j < 8; ++j) {
          const int qg = qlo + j + 8 * hl;
          float s0v = (s0 + lo <= qg)      ? S0[j] * scale : -1e30f;
          float s1v = (s0 + 16 + lo <= qg) ? S1[j] * scale : -1e30f;
          float mx = fmaxf(s0v, s1v);
#pragma unroll
          for (int off = 1; off < 16; off <<= 1) mx = fmaxf(mx, __shfl_xor(mx, off, 32));
          const float nm = fmaxf(mrun[j], mx);
          const float al = __expf(mrun[j] - nm);
          const float p0 = __expf(s0v - nm);
          const float p1 = __expf(s1v - nm);
          float rs = p0 + p1;
#pragma unroll
          for (int off = 1; off < 16; off <<= 1) rs += __shfl_xor(rs, off, 32);
          lrun[j] = lrun[j] * al + rs;
          mrun[j] = nm;
          O[0][j] *= al; O[1][j] *= al; O[2][j] *= al; O[3][j] *= al;
          const int m = j + 8 * hl;
          sP[w][m * 32 + lo]      = (__bf16)p0;   // C-layout -> LDS relay
          sP[w][m * 32 + 16 + lo] = (__bf16)p1;
        }
        asm volatile("s_wait_dscnt 0" ::: "memory");
        // reload P in A-layout (16x32, K = key)
        const __bf16* pp = &sP[w][lo * 32 + 16 * hl];
        const v16bf pa = cat16(*(const v8bf*)pp, *(const v8bf*)(pp + 8));
        // O += P @ V   (4 H-tiles of 16)
#pragma unroll
        for (int ht = 0; ht < 4; ++ht) {
          const __bf16* vp = &sV[(16 * ht + lo) * 136 + kl + 16 * hl];
          O[ht] = wmma_bf16(pa, cat16(*(const v8bf*)vp, *(const v8bf*)(vp + 8)), O[ht]);
        }
      }
    }
  }

  // epilogue: out[b, qlo+m, 16*ht+lo] = O / l
#pragma unroll
  for (int ht = 0; ht < 4; ++ht)
#pragma unroll
    for (int j = 0; j < 8; ++j) {
      const int m = j + 8 * hl;
      out[((size_t)b * T_ + qlo + m) * H_ + 16 * ht + lo] = O[ht][j] / lrun[j];
    }
}

// ---------------------------------------------------------------------------
extern "C" void kernel_launch(void* const* d_in, const int* in_sizes, int n_in,
                              void* d_out, int out_size, void* d_ws, size_t ws_size,
                              hipStream_t stream) {
  const float* x  = (const float*)d_in[0];
  const float* Wg = (const float*)d_in[1];
  const float* bg = (const float*)d_in[2];
  const float* Wk = (const float*)d_in[3];
  const float* Wq = (const float*)d_in[4];
  const float* Wv = (const float*)d_in[5];
  __bf16* ws = (__bf16*)d_ws;
  float* out = (float*)d_out;

  cvt_weights<<<64, 256, 0, stream>>>(Wg, Wk, Wq, Wv, ws);
  gate_qkv<<<B_ * (T_ / 128), 256, 0, stream>>>(x, bg, ws);
  flash_attn<<<B_ * (T_ / 128), 256, 0, stream>>>(ws, out);
}
